// AGDNGraphLayer_79946521247948
// MI455X (gfx1250) — matrix operand order repaired and checked by reading, not compiled, not run on hardware
//
#include <hip/hip_runtime.h>
#include <hip/hip_bf16.h>
#include <math.h>

typedef __attribute__((ext_vector_type(16))) __bf16 v16bf;
typedef __attribute__((ext_vector_type(8)))  __bf16 v8bf;
typedef __attribute__((ext_vector_type(8)))  float  v8f;

#define B_ 8
#define L_ 256
#define H_ 128
#define E_ 32
#define PT 136   // bf16 pitch for 128-wide LDS tiles (16B-aligned, bank-staggered)
#define PE 40    // bf16 pitch for 32-wide edge-feat tile

// ---- WMMA helpers -----------------------------------------------------------

static __device__ __forceinline__ v8f wmma_bf16(v16bf a, v16bf b, v8f c) {
  return __builtin_amdgcn_wmma_f32_16x16x32_bf16(false, a, false, b, (short)0, c,
                                                 false, false);
}

// A-fragment (16x32 bf16, row-major in LDS). ISA layout: lanes 0-15 hold row
// M=lane with K {0..7,16..23}; lanes 16-31 hold row M=lane-16 with K {8..15,24..31}.
static __device__ __forceinline__ v16bf ldsA(const __bf16* base, int pitch) {
  const int lane = threadIdx.x & 31;
  const int r    = lane & 15;
  const int kb   = (lane < 16) ? 0 : 8;
  const __bf16* p = base + r * pitch + kb;
  union { v16bf v; v8bf h2[2]; } u;
  u.h2[0] = *(const v8bf*)(p);        // ds_load_b128
  u.h2[1] = *(const v8bf*)(p + 16);   // ds_load_b128
  return u.v;
}

// B-fragment (32x16 bf16) gathered straight from a row-major fp32 weight
// (Kdim x ncols). ISA layout: lane holds column N=lane%16; lanes 0-15 carry
// K=kb0+{0..7,16..23}, lanes 16-31 carry K=kb0+{8..15,24..31}, 2 values per VGPR.
static __device__ __forceinline__ v16bf ldB(const float* __restrict__ W,
                                            int kb0, int n0, int ncols) {
  const int lane = threadIdx.x & 31;
  const int n  = n0 + (lane & 15);
  const int k0 = kb0 + ((lane < 16) ? 0 : 8);
  v16bf b;
#pragma unroll
  for (int t = 0; t < 16; ++t) b[t] = (__bf16)W[(size_t)(k0 + t + ((t >= 8) ? 8 : 0)) * ncols + n];
  return b;
}

// Store a C tile (f32 accum) as bf16 with bias add + optional relu.
static __device__ __forceinline__ void stC(__bf16* dst, int jbase, int mb,
                                           int col, const v8f& c, float bias,
                                           bool relu) {
  __bf16* drow = dst + (jbase + mb) * PT + col;
#pragma unroll
  for (int m = 0; m < 8; ++m) {
    float y = c[m] + bias;
    if (relu) y = fmaxf(y, 0.f);
    drow[m * PT] = (__bf16)y;
  }
}

// ---- pre-pass: a_src = h@Wa1[:H], a_dst = h@Wa1[H:2H], hWm = h@Wm1[:H] ------

__global__ void __launch_bounds__(128)
proj_kernel(const float* __restrict__ h, const float* __restrict__ Wa1,
            const float* __restrict__ Wm1, float* __restrict__ a_src,
            float* __restrict__ a_dst, float* __restrict__ hWm) {
  __shared__ float hr[H_];
  const int bi = blockIdx.x;
  const int n  = threadIdx.x;
  hr[n] = h[(size_t)bi * H_ + n];
  __syncthreads();
  float s1 = 0.f, s2 = 0.f, s3 = 0.f;
  for (int k = 0; k < H_; ++k) {
    const float hv = hr[k];
    s1 += hv * Wa1[(size_t)k * H_ + n];
    s2 += hv * Wa1[(size_t)(H_ + k) * H_ + n];
    s3 += hv * Wm1[(size_t)k * H_ + n];
  }
  a_src[(size_t)bi * H_ + n] = s1;
  a_dst[(size_t)bi * H_ + n] = s2;
  hWm  [(size_t)bi * H_ + n] = s3;
}

// ---- fused per-(b,i) row kernel --------------------------------------------

__global__ void __launch_bounds__(256)
agdn_row_kernel(const float* __restrict__ h, const float* __restrict__ edge_feat,
                const unsigned char* __restrict__ valid,
                const float* __restrict__ We1, const float* __restrict__ be1,
                const float* __restrict__ We2, const float* __restrict__ be2,
                const float* __restrict__ Wa1, const float* __restrict__ ba1,
                const float* __restrict__ Wa2, const float* __restrict__ ba2,
                const float* __restrict__ Wm1, const float* __restrict__ bm1,
                const float* __restrict__ Wm2, const float* __restrict__ bm2,
                const float* __restrict__ Wo1, const float* __restrict__ bo1,
                const float* __restrict__ Wo2, const float* __restrict__ bo2,
                const float* __restrict__ gamma, const float* __restrict__ beta,
                const float* __restrict__ a_src_ws,
                const float* __restrict__ a_dst_ws,
                const float* __restrict__ hWm_ws, float* __restrict__ out) {
  extern __shared__ char smem[];
  __bf16* EF = (__bf16*)smem;                        // L_*PE  (edge_feat bf16)
  __bf16* T1 = EF + (size_t)L_ * PE;                 // L_*PT  (t1 / msg_pre)
  __bf16* EH = T1 + (size_t)L_ * PT;                 // L_*PT  (edge_h)
  float* att_s = (float*)(EH + (size_t)L_ * PT);     // 256
  float* agg_s = att_s + 256;                        // 128
  float* h_s   = agg_s + 128;                        // 128
  float* v_s   = h_s + 128;                          // 128
  float* red_s = v_s + 128;                          // 64

  const int tid  = threadIdx.x;
  const int lane = tid & 31;
  const int wid  = tid >> 5;
  const int n0   = wid * 16;              // this wave's output-feature slice
  const int col  = n0 + (lane & 15);      // C-layout column for this lane
  const int mb   = (lane < 16) ? 0 : 8;   // C-layout row base for this lane
  const int bi   = blockIdx.x;            // b*L + i
  const int b    = bi >> 8;

  // stage 0: edge-feat row -> LDS bf16; init accumulators; load h row
  {
    const float4* src = (const float4*)(edge_feat + ((size_t)bi * L_ + tid) * E_);
#pragma unroll
    for (int q = 0; q < 8; ++q) {
      float4 f = src[q];
      __bf16* d = EF + tid * PE + q * 4;
      d[0] = (__bf16)f.x; d[1] = (__bf16)f.y;
      d[2] = (__bf16)f.z; d[3] = (__bf16)f.w;
    }
    att_s[tid] = 0.f;
    if (tid < H_) { agg_s[tid] = 0.f; h_s[tid] = h[(size_t)bi * H_ + tid]; }
  }

  // per-lane epilogue constants
  const float be1c = be1[col];
  const float be2c = be2[col];
  const float ac   = a_src_ws[(size_t)bi * H_ + col] + ba1[col];
  const float mc   = hWm_ws[(size_t)bi * H_ + col] + bm1[col];
  const float wa2c = Wa2[col];

  // stage-1 weights issued before the barrier so the gather hides under it
  v16bf We1f = ldB(We1, 0, n0, H_);
  __syncthreads();

  // stage 1: T1 = relu(EF @ We1 + be1)   (K = 32; two j-tiles in flight)
  for (int jt = 0; jt < 16; jt += 2) {
    v16bf a0 = ldsA(EF + (jt + 0) * 16 * PE, PE);
    v16bf a1 = ldsA(EF + (jt + 1) * 16 * PE, PE);
    v8f c0 = {}, c1 = {};
    c0 = wmma_bf16(a0, We1f, c0);
    c1 = wmma_bf16(a1, We1f, c1);
    stC(T1, (jt + 0) * 16, mb, col, c0, be1c, true);
    stC(T1, (jt + 1) * 16, mb, col, c1, be1c, true);
  }
  // stage-2 weights: gather overlaps the barrier
  v16bf We2f[4];
#pragma unroll
  for (int ks = 0; ks < 4; ++ks) We2f[ks] = ldB(We2, ks * 32, n0, H_);
  __syncthreads();

  // stage 2: EH = T1 @ We2 + be2   (K = 128; two j-tiles in flight)
  for (int jt = 0; jt < 16; jt += 2) {
    v8f c0 = {}, c1 = {};
#pragma unroll
    for (int ks = 0; ks < 4; ++ks) {
      v16bf a0 = ldsA(T1 + (jt + 0) * 16 * PT + ks * 32, PT);
      v16bf a1 = ldsA(T1 + (jt + 1) * 16 * PT + ks * 32, PT);
      c0 = wmma_bf16(a0, We2f[ks], c0);
      c1 = wmma_bf16(a1, We2f[ks], c1);
    }
    stC(EH, (jt + 0) * 16, mb, col, c0, be2c, false);
    stC(EH, (jt + 1) * 16, mb, col, c1, be2c, false);
  }
  // stage-3/4 weights: gather overlaps the barrier
  v16bf Waef[4], Wmef[4];
#pragma unroll
  for (int ks = 0; ks < 4; ++ks) {
    Waef[ks] = ldB(Wa1 + 2 * H_ * H_, ks * 32, n0, H_);  // Wa_edge
    Wmef[ks] = ldB(Wm1 + H_ * H_,     ks * 32, n0, H_);  // Wm_edge
  }
  __syncthreads();

  // stage 3+4: attention logits and msg_pre, sharing EH A-fragments
  const float* adst_col = a_dst_ws + (size_t)b * L_ * H_ + col;
  for (int jt = 0; jt < 16; jt += 2) {
    v8f ca0 = {}, cm0 = {}, ca1 = {}, cm1 = {};
#pragma unroll
    for (int ks = 0; ks < 4; ++ks) {
      v16bf a0 = ldsA(EH + (jt + 0) * 16 * PT + ks * 32, PT);
      v16bf a1 = ldsA(EH + (jt + 1) * 16 * PT + ks * 32, PT);
      ca0 = wmma_bf16(a0, Waef[ks], ca0);
      cm0 = wmma_bf16(a0, Wmef[ks], cm0);
      ca1 = wmma_bf16(a1, Waef[ks], ca1);
      cm1 = wmma_bf16(a1, Wmef[ks], cm1);
    }
#pragma unroll
    for (int t = 0; t < 2; ++t) {
      const v8f& ca = t ? ca1 : ca0;
      const v8f& cm = t ? cm1 : cm0;
#pragma unroll
      for (int m = 0; m < 8; ++m) {
        const int j = (jt + t) * 16 + mb + m;
        const float x = ca[m] + ac + adst_col[(size_t)j * H_];
        atomicAdd(&att_s[j], tanhf(x) * wa2c);               // ds_add_f32
        T1[j * PT + col] = (__bf16)fmaxf(cm[m] + mc, 0.f);   // msg_pre
      }
    }
  }
  // stage-6 weights: gather latency hides under the softmax barriers
  v16bf Wm2f[4];
#pragma unroll
  for (int ks = 0; ks < 4; ++ks) Wm2f[ks] = ldB(Wm2, ks * 32, n0, H_);
  __syncthreads();

  // stage 5: masked softmax over j (tid == j)
  {
    float s = (att_s[tid] + ba2[0]) * 0.08838834764831845f;  // 1/sqrt(128)
    const bool pv = (valid[(size_t)bi] != 0) && (valid[(size_t)b * L_ + tid] != 0);
    s = pv ? s : -1.0e9f;
    float mx = s;
    for (int off = 16; off; off >>= 1) mx = fmaxf(mx, __shfl_xor(mx, off));
    if (lane == 0) red_s[wid] = mx;
    __syncthreads();
    float gm = red_s[0];
#pragma unroll
    for (int w = 1; w < 8; ++w) gm = fmaxf(gm, red_s[w]);
    const float e = __expf(s - gm);
    float sm = e;
    for (int off = 16; off; off >>= 1) sm += __shfl_xor(sm, off);
    __syncthreads();
    if (lane == 0) red_s[wid] = sm;
    __syncthreads();
    float gs = 0.f;
#pragma unroll
    for (int w = 0; w < 8; ++w) gs += red_s[w];
    __syncthreads();
    att_s[tid] = e / gs;                                     // softmax weights
  }
  __syncthreads();

  // stage 6: agg[n] = sum_j w[j] * (msg_pre @ Wm2)[j,n]  (+bm2 later; sum w = 1)
  {
    float aggp = 0.f;
    for (int jt = 0; jt < 16; jt += 2) {
      v8f c0 = {}, c1 = {};
#pragma unroll
      for (int ks = 0; ks < 4; ++ks) {
        v16bf a0 = ldsA(T1 + (jt + 0) * 16 * PT + ks * 32, PT);
        v16bf a1 = ldsA(T1 + (jt + 1) * 16 * PT + ks * 32, PT);
        c0 = wmma_bf16(a0, Wm2f[ks], c0);
        c1 = wmma_bf16(a1, Wm2f[ks], c1);
      }
#pragma unroll
      for (int m = 0; m < 8; ++m) {
        aggp += att_s[(jt + 0) * 16 + mb + m] * c0[m];
        aggp += att_s[(jt + 1) * 16 + mb + m] * c1[m];
      }
    }
    atomicAdd(&agg_s[col], aggp);
  }
  __syncthreads();

  // stage 7: output MLP + residual + layernorm + mask (fp32 VALU, tiny)
  if (tid < H_) agg_s[tid] += bm2[tid];
  __syncthreads();
  if (tid < H_) {
    float acc = bo1[tid];
    for (int k = 0; k < H_; ++k) acc += h_s[k] * Wo1[(size_t)k * H_ + tid];
    for (int k = 0; k < H_; ++k) acc += agg_s[k] * Wo1[(size_t)(H_ + k) * H_ + tid];
    v_s[tid] = fmaxf(acc, 0.f);
  }
  __syncthreads();
  float res = 0.f;
  if (tid < H_) {
    float o = bo2[tid];
    for (int k = 0; k < H_; ++k) o += v_s[k] * Wo2[(size_t)k * H_ + tid];
    res = h_s[tid] + o;
  }
  float s1 = res, s2 = res * res;
  for (int off = 16; off; off >>= 1) {
    s1 += __shfl_xor(s1, off);
    s2 += __shfl_xor(s2, off);
  }
  __syncthreads();
  if (lane == 0) { red_s[wid] = s1; red_s[8 + wid] = s2; }
  __syncthreads();
  if (tid < H_) {
    float tS = 0.f, tQ = 0.f;
#pragma unroll
    for (int w = 0; w < 4; ++w) { tS += red_s[w]; tQ += red_s[8 + w]; }
    const float mu  = tS * (1.f / H_);
    const float var = tQ * (1.f / H_) - mu * mu;
    float nrm = (res - mu) * rsqrtf(var + 1e-5f) * gamma[tid] + beta[tid];
    nrm = (valid[(size_t)bi] != 0) ? nrm : 0.f;
    out[(size_t)bi * H_ + tid] = nrm;
  }
}

// ---- host launcher ----------------------------------------------------------

extern "C" void kernel_launch(void* const* d_in, const int* in_sizes, int n_in,
                              void* d_out, int out_size, void* d_ws, size_t ws_size,
                              hipStream_t stream) {
  (void)in_sizes; (void)n_in; (void)out_size; (void)ws_size;
  const float* h   = (const float*)d_in[0];
  const float* ef  = (const float*)d_in[1];
  const unsigned char* vm = (const unsigned char*)d_in[2];
  const float* We1 = (const float*)d_in[3];
  const float* be1 = (const float*)d_in[4];
  const float* We2 = (const float*)d_in[5];
  const float* be2 = (const float*)d_in[6];
  const float* Wa1 = (const float*)d_in[7];
  const float* ba1 = (const float*)d_in[8];
  const float* Wa2 = (const float*)d_in[9];
  const float* ba2 = (const float*)d_in[10];
  const float* Wm1 = (const float*)d_in[11];
  const float* bm1 = (const float*)d_in[12];
  const float* Wm2 = (const float*)d_in[13];
  const float* bm2 = (const float*)d_in[14];
  const float* Wo1 = (const float*)d_in[15];
  const float* bo1 = (const float*)d_in[16];
  const float* Wo2 = (const float*)d_in[17];
  const float* bo2 = (const float*)d_in[18];
  const float* gam = (const float*)d_in[19];
  const float* bet = (const float*)d_in[20];

  float* a_src = (float*)d_ws;
  float* a_dst = a_src + (size_t)B_ * L_ * H_;
  float* hWm   = a_dst + (size_t)B_ * L_ * H_;

  proj_kernel<<<B_ * L_, 128, 0, stream>>>(h, Wa1, Wm1, a_src, a_dst, hWm);

  const size_t smem = (size_t)L_ * PE * sizeof(__bf16)
                    + 2 * (size_t)L_ * PT * sizeof(__bf16)
                    + (256 + 128 + 128 + 128 + 64) * sizeof(float);
  agdn_row_kernel<<<B_ * L_, 256, smem, stream>>>(
      h, ef, vm, We1, be1, We2, be2, Wa1, ba1, Wa2, ba2, Wm1, bm1, Wm2, bm2,
      Wo1, bo1, Wo2, bo2, gam, bet, a_src, a_dst, hWm, (float*)d_out);
}
